// GRU_ngram_attn_51067161149600
// MI455X (gfx1250) — matrix operand order repaired and checked by reading, not compiled, not run on hardware
//
#include <hip/hip_runtime.h>
#if __has_include(<hip/amd_detail/amd_gfx1250_TDM.h>)
#define TDM_SIX_ARG 1
#endif

// ---------------------------------------------------------------------------
// Types for CDNA5 WMMA (wave32, 16x16x32 bf16 -> f32) and TDM descriptors
// ---------------------------------------------------------------------------
typedef __attribute__((ext_vector_type(16))) __bf16 v16bf;
typedef __attribute__((ext_vector_type(8)))  __bf16 v8bf;
typedef __attribute__((ext_vector_type(8)))  float  v8f;
typedef __attribute__((ext_vector_type(4)))  unsigned int v4u;
typedef __attribute__((ext_vector_type(8)))  int v8i;
typedef __attribute__((ext_vector_type(4)))  int v4i;

#define BS   512
#define SEQ  32
#define EDIM 1024
#define HDIM 1024
#define NG   3
#define NH   2
#define MID  30          // SEQ - NG + 1
#define KDIM 1024        // all GEMMs have K = 1024
#define NCAT 3072        // fused z|r|h gate width

#if defined(__gfx1250__) && __has_builtin(__builtin_amdgcn_tensor_load_to_lds)
#define HAVE_TDM 1
#else
#define HAVE_TDM 0
#endif

// f32 -> bf16 round-to-nearest-even (bit twiddle; no reliance on cvt builtins)
__device__ __forceinline__ __bf16 f2bf(float f) {
  union { float f; unsigned u; } v; v.f = f;
  unsigned r = v.u + 0x7FFFu + ((v.u >> 16) & 1u);
  unsigned short h = (unsigned short)(r >> 16);
  __bf16 out; __builtin_memcpy(&out, &h, 2); return out;
}
__device__ __forceinline__ float sigmoidf_(float v) { return 1.0f / (1.0f + expf(-v)); }

// ---------------------------------------------------------------------------
// WMMA fragment loaders from LDS (rows padded to 48 bf16 = 96B, 16B aligned)
// A-frag (16x32): lane L holds row M=L%16; K = {kb..kb+7} U {kb+16..kb+23}, kb=8*(L/16)
// B-frag (32x16): lane L holds col N=L%16; K = {kb..kb+7} U {kb+16..kb+23}, kb=8*(L/16)
// ---------------------------------------------------------------------------
__device__ __forceinline__ v16bf frag_from(const __bf16* p0, const __bf16* p1) {
  v8bf lo = *(const v8bf*)p0;
  v8bf hi = *(const v8bf*)p1;
  v16bf f;
#pragma unroll
  for (int i = 0; i < 8; ++i) { f[i] = lo[i]; f[i + 8] = hi[i]; }
  return f;
}
__device__ __forceinline__ v16bf load_frag_a(const __bf16* lds, int row) {
  const int lane = (int)(threadIdx.x & 31);
  const int r  = row + (lane & 15);
  const int kb = (lane >> 4) << 3;               // 0 or 8
  return frag_from(&lds[r * 48 + kb], &lds[r * 48 + kb + 16]);
}
__device__ __forceinline__ v16bf load_frag_b(const __bf16* lds, int row) {
  const int lane = (int)(threadIdx.x & 31);
  const int r  = row + (lane & 15);
  const int kb = (lane >> 4) << 3;               // 0 or 8
  return frag_from(&lds[r * 48 + kb], &lds[r * 48 + kb + 16]);
}

#if HAVE_TDM
// ---------------------------------------------------------------------------
// Tensor Data Mover: load one 128-row x 32-col bf16 tile (row stride 1024)
// into LDS with padding 64B-data + 32B-pad per row (=> 48 bf16 LDS rows).
// D# packing per CDNA5 ISA:
//   g0: count=1 | lds_addr | global_addr[56:0] | type=2
//   g1: data_size=2B, pad_enable, pad_interval=3 (16 DW), pad_amount=7 (8 DW),
//       tensor_dim0=1024, tensor_dim1=rows, tile_dim0=32, tile_dim1=128,
//       tensor_dim0_stride=1024
// ---------------------------------------------------------------------------
__device__ __forceinline__ unsigned lds_off(const void* p) {
  return (unsigned)(unsigned long long)(__SIZE_TYPE__)p;   // flat addr low 32b = LDS offset
}
__device__ __forceinline__ void tdm_load_tile(const __bf16* gsrc, unsigned ldsAddr,
                                              unsigned rows) {
  unsigned long long ga = (unsigned long long)(__SIZE_TYPE__)gsrc;
  v4u g0;
  g0[0] = 1u;                                              // count=1 (user mode)
  g0[1] = ldsAddr;                                         // lds_addr (bytes)
  g0[2] = (unsigned)ga;                                    // global_addr[31:0]
  g0[3] = (unsigned)((ga >> 32) & 0x01FFFFFFu) | (2u << 30); // addr[56:32] | type=2
  v8i g1;
  g1[0] = (int)((1u << 16) | (1u << 20) | (3u << 22) | (7u << 25));
  g1[1] = (int)((1024u & 0xFFFFu) << 16);                  // tensor_dim0 lo16
  g1[2] = (int)((1024u >> 16) | ((rows & 0xFFFFu) << 16)); // dim0 hi16 | dim1 lo16
  g1[3] = (int)((rows >> 16) | (32u << 16));               // dim1 hi16 | tile_dim0=32
  g1[4] = (int)128u;                                       // tile_dim1=128, tile_dim2=0
  g1[5] = (int)1024;                                       // tensor_dim0_stride lo32
  g1[6] = 0;
  g1[7] = 0;
  v4i gz = {0, 0, 0, 0};
#if defined(TDM_SIX_ARG)
  v8i gz8 = {0, 0, 0, 0, 0, 0, 0, 0};
  __builtin_amdgcn_tensor_load_to_lds(g0, g1, gz, gz, gz8, 0);
#else
  __builtin_amdgcn_tensor_load_to_lds(g0, g1, gz, gz, 0);
#endif
}
#endif  // HAVE_TDM

// ---------------------------------------------------------------------------
// C[M x N] f32 = A[M x 1024] bf16 (row-major) * B[N x 1024] bf16 (row-major)^T
// Block tile 128x128, 8 waves, wave tile 32x64 (2x4 frags of 16x16x32).
// Double-buffered LDS tiles filled by the Tensor Data Mover (wave 0 issues,
// workgroup syncs on TENSORcnt). M,N multiples of 128 guaranteed by caller.
// ---------------------------------------------------------------------------
__global__ __launch_bounds__(256) void gemm_bf16_wmma(
    const __bf16* __restrict__ A, const __bf16* __restrict__ B,
    float* __restrict__ C, int N, int M) {
  const int tid   = (int)threadIdx.x;
  const int mBase = (int)blockIdx.y * 128;
  const int nBase = (int)blockIdx.x * 128;

  __shared__ __bf16 lA[2][128 * 48];
  __shared__ __bf16 lB[2][128 * 48];

  const int wid = tid >> 5;
  const int mW  = (wid & 3) * 32;   // 4 waves along M
  const int nW  = (wid >> 2) * 64;  // 2 waves along N

  v8f acc[2][4] = {};

#if HAVE_TDM
  const __bf16* Atile = A + (size_t)mBase * KDIM;
  const __bf16* Btile = B + (size_t)nBase * KDIM;
  if (tid < 32) {                                   // wave 0 drives the TDM
    tdm_load_tile(Atile, lds_off(&lA[0][0]), (unsigned)M);
    tdm_load_tile(Btile, lds_off(&lB[0][0]), (unsigned)N);
  }
  for (int kk = 0; kk < KDIM / 32; ++kk) {
    const int cur = kk & 1;
    if (tid < 32) {
      if (kk + 1 < KDIM / 32) {                     // prefetch next K-tile pair
        tdm_load_tile(Atile + (size_t)(kk + 1) * 32, lds_off(&lA[cur ^ 1][0]), (unsigned)M);
        tdm_load_tile(Btile + (size_t)(kk + 1) * 32, lds_off(&lB[cur ^ 1][0]), (unsigned)N);
        __builtin_amdgcn_s_wait_tensorcnt(2);       // in-order => current pair done
      } else {
        __builtin_amdgcn_s_wait_tensorcnt(0);
      }
    }
    __syncthreads();

    v16bf aF[2], bF[4];
    aF[0] = load_frag_a(lA[cur], mW);
    aF[1] = load_frag_a(lA[cur], mW + 16);
#pragma unroll
    for (int j = 0; j < 4; ++j) bF[j] = load_frag_b(lB[cur], nW + j * 16);

#pragma unroll
    for (int i = 0; i < 2; ++i)
#pragma unroll
      for (int j = 0; j < 4; ++j)
        acc[i][j] = __builtin_amdgcn_wmma_f32_16x16x32_bf16(
            false, aF[i], false, bF[j], (short)0, acc[i][j], false, false);
    __syncthreads();
  }
#else
  // Fallback: VGPR-staged LDS tiles (single buffer)
  (void)M;
  for (int k0 = 0; k0 < KDIM; k0 += 32) {
#pragma unroll
    for (int cc = 0; cc < 2; ++cc) {
      const int c   = tid + cc * 256;
      const int row = c >> 2;
      const int col = (c & 3) << 3;
      *(v8bf*)&lA[0][row * 48 + col] =
          *(const v8bf*)&A[(size_t)(mBase + row) * KDIM + k0 + col];
      *(v8bf*)&lB[0][row * 48 + col] =
          *(const v8bf*)&B[(size_t)(nBase + row) * KDIM + k0 + col];
    }
    if (k0 + 32 < KDIM) {
      __builtin_prefetch(&A[(size_t)(mBase + (tid & 127)) * KDIM + k0 + 32], 0, 0);
      __builtin_prefetch(&B[(size_t)(nBase + (tid & 127)) * KDIM + k0 + 32], 0, 0);
    }
    __syncthreads();
    v16bf aF[2], bF[4];
    aF[0] = load_frag_a(lA[0], mW);
    aF[1] = load_frag_a(lA[0], mW + 16);
#pragma unroll
    for (int j = 0; j < 4; ++j) bF[j] = load_frag_b(lB[0], nW + j * 16);
#pragma unroll
    for (int i = 0; i < 2; ++i)
#pragma unroll
      for (int j = 0; j < 4; ++j)
        acc[i][j] = __builtin_amdgcn_wmma_f32_16x16x32_bf16(
            false, aF[i], false, bF[j], (short)0, acc[i][j], false, false);
    __syncthreads();
  }
#endif

  // C/D layout: VGPR r, lanes 0-15 -> M=r, lanes 16-31 -> M=r+8; N = lane%16
  const int lane = tid & 31;
  const int cn   = nBase + nW + (lane & 15);
  const int mTop = mBase + mW + ((lane >> 4) << 3);
#pragma unroll
  for (int i = 0; i < 2; ++i)
#pragma unroll
    for (int j = 0; j < 4; ++j)
#pragma unroll
      for (int r = 0; r < 8; ++r)
        C[(size_t)(mTop + i * 16 + r) * N + cn + j * 16] = acc[i][j][r];
}

// ---------------------------------------------------------------------------
// f32 -> bf16 copy (weight packing into concatenated z|r|h layout)
// ---------------------------------------------------------------------------
__global__ __launch_bounds__(256) void conv_bf16_kernel(
    const float* __restrict__ src, __bf16* __restrict__ dst, int n) {
  int i = (int)(blockIdx.x * 256 + threadIdx.x);
  if (i < n) dst[i] = f2bf(src[i]);
}

// ---------------------------------------------------------------------------
// xs[dir][b][m][e] = sum_{g<3} x_dir[b][m+g][e]  (bf16 output, both dirs)
// ---------------------------------------------------------------------------
__global__ __launch_bounds__(256) void xs_kernel(
    const float* __restrict__ x, __bf16* __restrict__ xs) {
  unsigned i = blockIdx.x * 256 + threadIdx.x;   // < 2*512*30*1024
  int e = i & 1023;
  unsigned r = i >> 10;
  int m = (int)(r % MID);
  unsigned rb = r / MID;
  int b = (int)(rb & (BS - 1));
  int dir = (int)(rb >> 9);
  int s0 = dir ? (31 - m) : m;
  int s1 = dir ? (30 - m) : (m + 1);
  int s2 = dir ? (29 - m) : (m + 2);
  const float* xb = x + (size_t)b * SEQ * EDIM;
  float v = xb[(size_t)s0 * EDIM + e] + xb[(size_t)s1 * EDIM + e] +
            xb[(size_t)s2 * EDIM + e];
  xs[i] = f2bf(v);
}

// ---------------------------------------------------------------------------
// ud[dir][b][s][h] = dot(x_dir[b][s], Ua[dir][h])   (h = 0,1)
// one 128-thread block per (dir,b,s)
// ---------------------------------------------------------------------------
__global__ __launch_bounds__(128) void ud_kernel(
    const float* __restrict__ x, const float* __restrict__ Ua,
    float* __restrict__ ud) {
  int id  = (int)blockIdx.x;            // < 2*512*32
  int s   = id & 31;
  int b   = (id >> 5) & 511;
  int dir = id >> 14;
  int tid = (int)threadIdx.x;
  int srow = dir ? (31 - s) : s;
  const float* xr = x + ((size_t)b * SEQ + srow) * EDIM;
  const float* u  = Ua + (size_t)dir * NH * EDIM;
  float p0 = 0.f, p1 = 0.f;
  for (int e = tid; e < EDIM; e += 128) {
    float xv = xr[e];
    p0 += xv * u[e];
    p1 += xv * u[EDIM + e];
  }
  __shared__ float s0[128], s1[128];
  s0[tid] = p0; s1[tid] = p1; __syncthreads();
  for (int st = 64; st > 0; st >>= 1) {
    if (tid < st) { s0[tid] += s0[tid + st]; s1[tid] += s1[tid + st]; }
    __syncthreads();
  }
  if (tid == 0) {
    size_t base = ((size_t)id) * 2;     // [(dir*512+b)*32 + s]*2
    ud[base] = s0[0]; ud[base + 1] = s1[0];
  }
}

// ---------------------------------------------------------------------------
// ht init: ht = h0[dir], ht_bf = bf16(h0[dir])
// ---------------------------------------------------------------------------
__global__ __launch_bounds__(256) void init_ht_kernel(
    const float* __restrict__ h0, float* __restrict__ ht,
    __bf16* __restrict__ htbf) {
  int i = (int)(blockIdx.x * 256 + threadIdx.x);   // < 512*1024
  float v = h0[i];
  ht[i] = v;
  htbf[i] = f2bf(v);
}

// ---------------------------------------------------------------------------
// attention scalars: alpha[b][g] from ht, Wa_w, Wa_b, fh_w, fh_b, ud
// one 256-thread block per batch element
// ---------------------------------------------------------------------------
__global__ __launch_bounds__(256) void attn_kernel(
    const float* __restrict__ ht, const float* __restrict__ Wa_w,
    const float* __restrict__ Wa_b, const float* __restrict__ fh_w,
    const float* __restrict__ fh_b, const float* __restrict__ ud,
    int t, float* __restrict__ alpha) {
  int b = (int)blockIdx.x;
  int tid = (int)threadIdx.x;
  float p[6] = {0.f, 0.f, 0.f, 0.f, 0.f, 0.f};   // j = h*3 + g
  const float* hb = ht + (size_t)b * HDIM;
  for (int d = tid; d < HDIM; d += 256) {
    float hv = hb[d];
#pragma unroll
    for (int j = 0; j < 6; ++j) p[j] += hv * Wa_w[(size_t)j * HDIM + d];
  }
  __shared__ float sb[256];
  __shared__ float wa[6];
#pragma unroll
  for (int j = 0; j < 6; ++j) {
    sb[tid] = p[j]; __syncthreads();
    for (int st = 128; st > 0; st >>= 1) {
      if (tid < st) sb[tid] += sb[tid + st];
      __syncthreads();
    }
    if (tid == 0) wa[j] = sb[0];
    __syncthreads();
  }
  if (tid == 0) {
    float fw0 = fh_w[0], fw1 = fh_w[1], fbb = fh_b[0];
    float e[3];
#pragma unroll
    for (int g = 0; g < 3; ++g) {
      float u0 = ud[((size_t)b * SEQ + t + g) * 2 + 0] + wa[g]     + Wa_b[g];
      float u1 = ud[((size_t)b * SEQ + t + g) * 2 + 1] + wa[3 + g] + Wa_b[3 + g];
      e[g] = tanhf(u0 * fw0 + u1 * fw1 + fbb);
    }
    float m = fmaxf(e[0], fmaxf(e[1], e[2]));
    float x0 = expf(e[0] - m), x1 = expf(e[1] - m), x2 = expf(e[2] - m);
    float inv = 1.0f / (x0 + x1 + x2);
    alpha[b * 3 + 0] = x0 * inv;
    alpha[b * 3 + 1] = x1 * inv;
    alpha[b * 3 + 2] = x2 * inv;
  }
}

// ---------------------------------------------------------------------------
// fused GRU update + attention readout + ht bf16 repack
// G: [512][3072] = ht @ [Uz|Ur|Uh]^T ;  P: [b*30+t][3072] precomputed x-part
// ---------------------------------------------------------------------------
__global__ __launch_bounds__(256) void update_kernel(
    float* __restrict__ ht, __bf16* __restrict__ htbf,
    const float* __restrict__ G, const float* __restrict__ P,
    const float* __restrict__ x, const float* __restrict__ alpha,
    float* __restrict__ X, int t, int rev) {
  int i = (int)(blockIdx.x * 256 + threadIdx.x);   // < 512*1024
  int b = i >> 10, d = i & 1023;
  size_t prow = ((size_t)b * MID + t) * NCAT;
  size_t grow = (size_t)b * NCAT;
  float z  = sigmoidf_(P[prow + d]        + G[grow + d]);
  float r  = sigmoidf_(P[prow + 1024 + d] + G[grow + 1024 + d]);
  float hw = tanhf   (P[prow + 2048 + d]  + r * G[grow + 2048 + d]);
  float hp = ht[i];
  float hn = (1.0f - z) * hp + z * hw;
  ht[i]   = hn;
  htbf[i] = f2bf(hn);

  float a0 = alpha[b * 3 + 0], a1 = alpha[b * 3 + 1], a2 = alpha[b * 3 + 2];
  int s0 = rev ? (31 - t) : t;
  int s1 = rev ? (30 - t) : (t + 1);
  int s2 = rev ? (29 - t) : (t + 2);
  const float* xb = x + (size_t)b * SEQ * EDIM;
  float o = a0 * xb[(size_t)s0 * EDIM + d] + a1 * xb[(size_t)s1 * EDIM + d] +
            a2 * xb[(size_t)s2 * EDIM + d];
  X[((size_t)b * MID + t) * EDIM + d] = o;
}

// ---------------------------------------------------------------------------
// final projection: out[b][j][e] = fb_b[j] + sum_k fb_w[j][k] * Xb[b][k][e]
// Xb = [X0 (k<30) | X1 (k>=30)]
// ---------------------------------------------------------------------------
__global__ __launch_bounds__(256) void proj_kernel(
    const float* __restrict__ X0, const float* __restrict__ X1,
    const float* __restrict__ fb_w, const float* __restrict__ fb_b,
    float* __restrict__ out) {
  int i = (int)(blockIdx.x * 256 + threadIdx.x);   // < 512*1024
  int b = i >> 10, e = i & 1023;
  float acc[30];
#pragma unroll
  for (int j = 0; j < 30; ++j) acc[j] = fb_b[j];
  for (int k = 0; k < 30; ++k) {
    float v = X0[((size_t)b * MID + k) * EDIM + e];
#pragma unroll
    for (int j = 0; j < 30; ++j) acc[j] += fb_w[j * 60 + k] * v;
  }
  for (int k = 0; k < 30; ++k) {
    float v = X1[((size_t)b * MID + k) * EDIM + e];
#pragma unroll
    for (int j = 0; j < 30; ++j) acc[j] += fb_w[j * 60 + 30 + k] * v;
  }
#pragma unroll
  for (int j = 0; j < 30; ++j)
    out[((size_t)b * MID + j) * EDIM + e] = acc[j];
}

// ---------------------------------------------------------------------------
// host orchestration
// ---------------------------------------------------------------------------
extern "C" void kernel_launch(void* const* d_in, const int* in_sizes, int n_in,
                              void* d_out, int out_size, void* d_ws, size_t ws_size,
                              hipStream_t stream) {
  (void)in_sizes; (void)n_in; (void)out_size; (void)ws_size;
  const float* x    = (const float*)d_in[0];
  const float* h0   = (const float*)d_in[1];
  const float* Wsrc[3] = {(const float*)d_in[2], (const float*)d_in[4], (const float*)d_in[6]};
  const float* Usrc[3] = {(const float*)d_in[3], (const float*)d_in[5], (const float*)d_in[7]};
  const float* Ua   = (const float*)d_in[8];
  const float* Wa_w = (const float*)d_in[9];
  const float* Wa_b = (const float*)d_in[10];
  const float* fh_w = (const float*)d_in[11];
  const float* fh_b = (const float*)d_in[12];
  const float* fb_w = (const float*)d_in[13];
  const float* fb_b = (const float*)d_in[14];
  float* out = (float*)d_out;

  char* wsp = (char*)d_ws;
  auto carve = [&](size_t bytes) -> char* {
    char* r = wsp; wsp += (bytes + 255) & ~(size_t)255; return r;
  };
  const size_t MW = (size_t)HDIM * KDIM;             // 1M elements per weight matrix
  __bf16* wcat = (__bf16*)carve(2 * 3 * MW * 2);     // [dir][z|r|h][1024][1024]
  __bf16* ucat = (__bf16*)carve(2 * 3 * MW * 2);
  __bf16* xs   = (__bf16*)carve((size_t)2 * BS * MID * EDIM * 2);
  float*  P    = (float*) carve((size_t)2 * BS * MID * NCAT * 4);
  float*  ud   = (float*) carve((size_t)2 * BS * SEQ * NH * 4);
  float*  X0   = (float*) carve((size_t)BS * MID * EDIM * 4);
  float*  X1   = (float*) carve((size_t)BS * MID * EDIM * 4);
  float*  ht   = (float*) carve((size_t)BS * HDIM * 4);
  __bf16* htbf = (__bf16*)carve((size_t)BS * HDIM * 2);
  float*  G    = (float*) carve((size_t)BS * NCAT * 4);
  float*  alph = (float*) carve((size_t)BS * NG * 4);

  // 1) pack weights to bf16, gates concatenated z|r|h per direction
  for (int dir = 0; dir < 2; ++dir)
    for (int g = 0; g < 3; ++g) {
      conv_bf16_kernel<<<(int)(MW / 256), 256, 0, stream>>>(
          Wsrc[g] + (size_t)dir * MW, wcat + ((size_t)dir * 3 + g) * MW, (int)MW);
      conv_bf16_kernel<<<(int)(MW / 256), 256, 0, stream>>>(
          Usrc[g] + (size_t)dir * MW, ucat + ((size_t)dir * 3 + g) * MW, (int)MW);
    }

  // 2) window sums (bf16) + attention x-projections
  xs_kernel<<<(2 * BS * MID * EDIM) / 256, 256, 0, stream>>>(x, xs);
  ud_kernel<<<2 * BS * SEQ, 128, 0, stream>>>(x, Ua, ud);

  // 3) big parallel GEMMs: P[dir] = xs[dir] @ [Wz|Wr|Wh]^T   (15360 x 3072)
  for (int dir = 0; dir < 2; ++dir) {
    dim3 grid(NCAT / 128, (BS * MID) / 128);
    gemm_bf16_wmma<<<grid, 256, 0, stream>>>(
        xs + (size_t)dir * BS * MID * EDIM,
        wcat + (size_t)dir * 3 * MW,
        P + (size_t)dir * BS * MID * NCAT, NCAT, BS * MID);
  }

  // 4) serial scans
  for (int dir = 0; dir < 2; ++dir) {
    init_ht_kernel<<<(BS * HDIM) / 256, 256, 0, stream>>>(
        h0 + (size_t)dir * BS * HDIM, ht, htbf);
    float* Xout = (dir == 0) ? X0 : X1;
    for (int t = 0; t < MID; ++t) {
      dim3 grid(NCAT / 128, BS / 128);
      gemm_bf16_wmma<<<grid, 256, 0, stream>>>(
          htbf, ucat + (size_t)dir * 3 * MW, G, NCAT, BS);
      attn_kernel<<<BS, 256, 0, stream>>>(
          ht, Wa_w + (size_t)dir * NH * NG * HDIM, Wa_b + (size_t)dir * NH * NG,
          fh_w + (size_t)dir * NH, fh_b + dir,
          ud + (size_t)dir * BS * SEQ * NH, t, alph);
      update_kernel<<<(BS * HDIM) / 256, 256, 0, stream>>>(
          ht, htbf, G, P + (size_t)dir * BS * MID * NCAT, x, alph, Xout, t, dir);
    }
  }

  // 5) final 60 -> 30 projection
  proj_kernel<<<(BS * EDIM) / 256, 256, 0, stream>>>(X0, X1, fb_w, fb_b, out);
}
